// BottleneckAttn_10187662426317
// MI455X (gfx1250) — compile-verified, matrix operand order checked by reading, not run on hardware
//
#include <hip/hip_runtime.h>
#include <hip/hip_bf16.h>
#include <stdint.h>

#define DEVINL __device__ __forceinline__

typedef __attribute__((ext_vector_type(16))) __bf16        v16bf;
typedef __attribute__((ext_vector_type(8)))  float         v8f;
typedef __attribute__((ext_vector_type(4)))  unsigned int  u32x4;
typedef __attribute__((ext_vector_type(4)))  int           i32x4;
typedef __attribute__((ext_vector_type(8)))  int           i32x8;

union BF16Tile { u32x4 q[2]; v16bf v; };

static constexpr int BATCH = 16;
static constexpr int CC    = 512;
static constexpr int HW    = 1024;   // 32*32
static constexpr int NH    = 4;
static constexpr int DH    = 128;
static constexpr float SCALE = 0.08838834764831845f;  // 128^-0.5

#if defined(__has_builtin)
#if __has_builtin(__builtin_amdgcn_tensor_load_to_lds)
#define HAVE_TDM 1
#endif
#endif

// native bf16 convert (v_cvt_*bf16_f32, RNE) instead of manual bit twiddling
DEVINL unsigned short f2bf(float f) {
  __bf16 h = (__bf16)f;
  return __builtin_bit_cast(unsigned short, h);
}
DEVINL float bf2f(unsigned short h) {
  return __uint_as_float(((unsigned)h) << 16);
}

// ---------------------------------------------------------------------------
// Kernel 1: QKV projection  qkv[b,o,p] = sum_c w[o,c] * x[b,c,p]
// Writes: q_bf[b,n,p,d], k_bf[b,n,p,d] (row-major over d), v_t[b,n,d,p].
// ---------------------------------------------------------------------------
__launch_bounds__(256)
__global__ void qkv_proj_kernel(const float* __restrict__ x,
                                const float* __restrict__ w,
                                unsigned short* __restrict__ q_bf,
                                unsigned short* __restrict__ k_bf,
                                unsigned short* __restrict__ v_t) {
  __shared__ alignas(16) unsigned short wLds[16 * 40];    // 16 o-rows x 32 c (pad 40)
  __shared__ alignas(16) unsigned short xT[128 * 40];     // 128 p-rows x 32 c (pad 40)

  const int t    = threadIdx.x;
  const int lane = t & 31, wv = t >> 5;          // 8 waves
  const int lrow = lane & 15, half = lane >> 4, kbase = half * 8;
  const int p0 = blockIdx.x * 128;
  const int o0 = blockIdx.y * 16;
  const int b  = blockIdx.z;
  const float* xb = x + (size_t)b * CC * HW;

  v8f acc = {0.f, 0.f, 0.f, 0.f, 0.f, 0.f, 0.f, 0.f};

  for (int c0 = 0; c0 < CC; c0 += 32) {
    __syncthreads();
    // stage w tile (16 x 32) -> bf16
    {
      int idx = t;
      #pragma unroll
      for (int i = 0; i < 2; ++i) {
        int orow = idx >> 5, cl = idx & 31;
        wLds[orow * 40 + cl] = f2bf(w[(size_t)(o0 + orow) * CC + c0 + cl]);
        idx += 256;
      }
    }
    // stage x tile (32 c x 128 p), transposed to [p][c] bf16
    {
      int idx = t;
      #pragma unroll
      for (int i = 0; i < 16; ++i) {
        int cl = idx >> 7, pl = idx & 127;
        xT[pl * 40 + cl] = f2bf(xb[(size_t)(c0 + cl) * HW + p0 + pl]);
        idx += 256;
      }
    }
    __syncthreads();

    BF16Tile A, Bm;
    const unsigned short* ap = &wLds[lrow * 40 + kbase];
    A.q[0] = *(const u32x4*)(ap);
    A.q[1] = *(const u32x4*)(ap + 16);
    const unsigned short* bp = &xT[(wv * 16 + lrow) * 40 + kbase];
    Bm.q[0] = *(const u32x4*)(bp);
    Bm.q[1] = *(const u32x4*)(bp + 16);
    acc = __builtin_amdgcn_wmma_f32_16x16x32_bf16(false, A.v, false, Bm.v,
                                                  (short)0, acc, false, false);
  }

  // epilogue: route by o block (q / k / v)
  const int g = o0 >> 7, kind = g >> 2, n = g & 3, d0 = o0 & 127;
  const int bn = b * NH + n;
  const int p  = p0 + wv * 16 + lrow;

  if (kind == 2) {
    // V transposed: v_t[bn][d][p]
    #pragma unroll
    for (int i = 0; i < 8; ++i) {
      int d = d0 + half * 8 + i;
      v_t[((size_t)bn * DH + d) * HW + p] = f2bf(acc[i]);
    }
  } else {
    unsigned short* dst = (kind == 0) ? q_bf : k_bf;
    // transposed pack: lane owns (p, d0+half*8 .. +7) contiguous -> one b128 store
    union { unsigned short h[8]; u32x4 q; } pk;
    #pragma unroll
    for (int i = 0; i < 8; ++i) pk.h[i] = f2bf(acc[i]);
    *(u32x4*)(dst + ((size_t)bn * HW + p) * DH + d0 + half * 8) = pk.q;
  }
}

// ---------------------------------------------------------------------------
// Kernel 2: relative logits  qw[bn,p,r] = sum_d q[bn,p,d]*width_rel[r,d]
//           (r padded to 64; row 63 is zero)   qh likewise with height_rel.
// ---------------------------------------------------------------------------
__launch_bounds__(256)
__global__ void rel_logits_kernel(const unsigned short* __restrict__ q_bf,
                                  const float* __restrict__ height_rel,
                                  const float* __restrict__ width_rel,
                                  unsigned short* __restrict__ qw,
                                  unsigned short* __restrict__ qh) {
  __shared__ alignas(16) unsigned short relW[64 * 136];
  __shared__ alignas(16) unsigned short relH[64 * 136];

  const int t    = threadIdx.x;
  const int lane = t & 31, wv = t >> 5;
  const int lrow = lane & 15, half = lane >> 4, kbase = half * 8;
  const int bn    = blockIdx.y;
  const int pbase = blockIdx.x * 128 + wv * 16;

  for (int idx = t; idx < 64 * 128; idx += 256) {
    int r = idx >> 7, d = idx & 127;
    float fw = (r < 63) ? width_rel[(size_t)r * DH + d] : 0.f;
    float fh = (r < 63) ? height_rel[(size_t)r * DH + d] : 0.f;
    relW[r * 136 + d] = f2bf(fw);
    relH[r * 136 + d] = f2bf(fh);
  }
  __syncthreads();

  BF16Tile qa[4];
  const unsigned short* qrow = q_bf + ((size_t)bn * HW + pbase + lrow) * DH;
  #pragma unroll
  for (int s = 0; s < 4; ++s) {
    qa[s].q[0] = *(const u32x4*)(qrow + s * 32 + kbase);
    qa[s].q[1] = *(const u32x4*)(qrow + s * 32 + kbase + 16);
  }

  #pragma unroll
  for (int tab = 0; tab < 2; ++tab) {
    const unsigned short* rel = tab ? relH : relW;
    unsigned short* outp      = tab ? qh : qw;
    #pragma unroll
    for (int rt = 0; rt < 4; ++rt) {
      v8f acc = {0.f, 0.f, 0.f, 0.f, 0.f, 0.f, 0.f, 0.f};
      #pragma unroll
      for (int s = 0; s < 4; ++s) {
        BF16Tile Bm;
        const unsigned short* bp = rel + (rt * 16 + lrow) * 136 + s * 32 + kbase;
        Bm.q[0] = *(const u32x4*)bp;
        Bm.q[1] = *(const u32x4*)(bp + 16);
        acc = __builtin_amdgcn_wmma_f32_16x16x32_bf16(false, qa[s].v, false, Bm.v,
                                                      (short)0, acc, false, false);
      }
      #pragma unroll
      for (int i = 0; i < 8; ++i) {
        int m = half * 8 + i;
        outp[((size_t)bn * HW + pbase + m) * 64 + rt * 16 + lrow] = f2bf(acc[i]);
      }
    }
  }
}

// ---------------------------------------------------------------------------
// Kernel 3: flash attention with relative-position bias.
// Block = 128 threads (4 waves), each wave owns 16 query rows; 16 key chunks
// of 64.  K / V^T chunks are DMA'd into LDS by the Tensor Data Mover (with
// LDS padding reproducing the conflict-free strides); online softmax;
// O = sum P*V; output layout [b][p][n][d].
// ---------------------------------------------------------------------------
__launch_bounds__(128)
__global__ void attn_kernel(const unsigned short* __restrict__ q_bf,
                            const unsigned short* __restrict__ k_bf,
                            const unsigned short* __restrict__ v_t,
                            const unsigned short* __restrict__ qw,
                            const unsigned short* __restrict__ qh,
                            float* __restrict__ out) {
  __shared__ alignas(16) unsigned short kTile[64 * 136];    // 17408 B (128 halfs + 4-dword pad per row)
  __shared__ alignas(16) unsigned short vTile[128 * 72];    // 18432 B (64 halfs + 4-dword pad per row)
  __shared__ alignas(16) unsigned short pTile[4 * 16 * 72]; //  9216 B
  __shared__ alignas(16) unsigned short bW[4 * 16 * 64];    //  8192 B
  __shared__ alignas(16) unsigned short bH[4 * 16 * 64];    //  8192 B

  const int t    = threadIdx.x;
  const int lane = t & 31, wv = t >> 5;          // 4 waves
  const int lrow = lane & 15, half = lane >> 4, kbase = half * 8;
  const int bn    = blockIdx.y;
  const int qbase = blockIdx.x * 64 + wv * 16;

  // stage this wave's bias rows (wave-local; no block barrier needed)
  {
    const u32x4* sW = (const u32x4*)(qw + ((size_t)bn * HW + qbase) * 64);
    const u32x4* sH = (const u32x4*)(qh + ((size_t)bn * HW + qbase) * 64);
    u32x4* dW = (u32x4*)(bW + wv * 16 * 64);
    u32x4* dH = (u32x4*)(bH + wv * 16 * 64);
    for (int idx = lane; idx < 128; idx += 32) {   // 16*64/8 uint4
      dW[idx] = sW[idx];
      dH[idx] = sH[idx];
    }
  }

  // load Q A-tiles (16 rows x 128 d, 4 K-steps)
  BF16Tile qa[4];
  const unsigned short* qrow = q_bf + ((size_t)bn * HW + qbase + lrow) * DH;
  #pragma unroll
  for (int s = 0; s < 4; ++s) {
    qa[s].q[0] = *(const u32x4*)(qrow + s * 32 + kbase);
    qa[s].q[1] = *(const u32x4*)(qrow + s * 32 + kbase + 16);
  }

  int hq[8], wq[8];
  #pragma unroll
  for (int i = 0; i < 8; ++i) {
    int p = qbase + half * 8 + i;
    hq[i] = (p >> 5) & 31;
    wq[i] = p & 31;
  }

  v8f Oacc[8];
  #pragma unroll
  for (int dt = 0; dt < 8; ++dt) Oacc[dt] = (v8f){0.f,0.f,0.f,0.f,0.f,0.f,0.f,0.f};
  float rmax[8], rsum[8];
  #pragma unroll
  for (int i = 0; i < 8; ++i) { rmax[i] = -3.0e38f; rsum[i] = 0.f; }

  const unsigned short* kb = k_bf + (size_t)bn * HW * DH;
  const unsigned short* vb = v_t + (size_t)bn * DH * HW;
  const unsigned short* mW = bW + wv * 16 * 64;
  const unsigned short* mH = bH + wv * 16 * 64;
  unsigned short* pt = pTile + wv * 16 * 72;

#ifdef HAVE_TDM
  const unsigned ldsK = (unsigned)(size_t)(void*)kTile;   // LDS offset = low 32 bits of flat addr
  const unsigned ldsV = (unsigned)(size_t)(void*)vTile;
  // D# group1 (constant per kernel): data_size=2B, pad_enable,
  //   K: 64 rows x 128 elems, row stride 128; pad 4 dwords every 64 dwords (code 5,3)
  //   V: 128 rows x 64 elems, row stride 1024; pad 4 dwords every 32 dwords (code 4,3)
  const i32x8 g1k = { (int)0x07510000u, (int)(128u << 16), (int)(1024u << 16),
                      (int)(128u << 16), 64, 128, 0, 0 };
  const i32x8 g1v = { (int)0x07110000u, (int)(1024u << 16), (int)(128u << 16),
                      (int)(64u << 16), 128, 1024, 0, 0 };
  const i32x4 gz = {0, 0, 0, 0};
#endif

  for (int kc = 0; kc < 16; ++kc) {
    const int kp0 = kc * 64;
    __syncthreads();
#ifdef HAVE_TDM
    if (wv == 0) {
      unsigned long long gaK = (unsigned long long)(size_t)(kb + (size_t)kp0 * DH);
      unsigned long long gaV = (unsigned long long)(size_t)(vb + kp0);
      u32x4 g0k = { 1u, ldsK, (unsigned)gaK,
                    (unsigned)((gaK >> 32) & 0x01FFFFFFull) | 0x80000000u };  // count=1, type=2
      u32x4 g0v = { 1u, ldsV, (unsigned)gaV,
                    (unsigned)((gaV >> 32) & 0x01FFFFFFull) | 0x80000000u };
#if __clang_major__ >= 23
      i32x8 gz8 = {0, 0, 0, 0, 0, 0, 0, 0};
      __builtin_amdgcn_tensor_load_to_lds(g0k, g1k, gz, gz, gz8, 0);
      __builtin_amdgcn_tensor_load_to_lds(g0v, g1v, gz, gz, gz8, 0);
#else
      __builtin_amdgcn_tensor_load_to_lds(g0k, g1k, gz, gz, 0);
      __builtin_amdgcn_tensor_load_to_lds(g0v, g1v, gz, gz, 0);
#endif
      __builtin_amdgcn_s_wait_tensorcnt(0);
    }
#else
    // manual staging fallback
    for (int idx = t; idx < 1024; idx += 128) {
      int r = idx >> 4, cq = idx & 15;
      *(u32x4*)(kTile + r * 136 + cq * 8) =
          *(const u32x4*)(kb + (size_t)(kp0 + r) * DH + cq * 8);
    }
    for (int idx = t; idx < 1024; idx += 128) {
      int d = idx >> 3, cq = idx & 7;
      *(u32x4*)(vTile + d * 72 + cq * 8) =
          *(const u32x4*)(vb + (size_t)d * HW + kp0 + cq * 8);
    }
#endif
    if (kc + 1 < 16)  // pull next K chunk toward the caches (global_prefetch_b8)
      __builtin_prefetch(kb + (size_t)(kp0 + 64) * DH + t * 64, 0, 0);
    __syncthreads();

    // S = Q * K^T   (4 N-tiles of 16 keys)
    v8f sc[4];
    #pragma unroll
    for (int nt = 0; nt < 4; ++nt) {
      v8f a = {0.f,0.f,0.f,0.f,0.f,0.f,0.f,0.f};
      #pragma unroll
      for (int s = 0; s < 4; ++s) {
        BF16Tile Bm;
        const unsigned short* bp = kTile + (nt * 16 + lrow) * 136 + s * 32 + kbase;
        Bm.q[0] = *(const u32x4*)bp;
        Bm.q[1] = *(const u32x4*)(bp + 16);
        a = __builtin_amdgcn_wmma_f32_16x16x32_bf16(false, qa[s].v, false, Bm.v,
                                                    (short)0, a, false, false);
      }
      sc[nt] = a;
    }

    // scale + relative-position bias
    float chmax[8];
    #pragma unroll
    for (int i = 0; i < 8; ++i) chmax[i] = -3.0e38f;
    #pragma unroll
    for (int nt = 0; nt < 4; ++nt) {
      int kp = kp0 + nt * 16 + lrow;
      int h2 = (kp >> 5) & 31, w2 = kp & 31;
      #pragma unroll
      for (int i = 0; i < 8; ++i) {
        int m = half * 8 + i;
        float bwv = bf2f(mW[m * 64 + (w2 - wq[i] + 31)]);
        float bhv = bf2f(mH[m * 64 + (h2 - hq[i] + 31)]);
        float sv = sc[nt][i] * SCALE + bwv + bhv;
        sc[nt][i] = sv;
        chmax[i] = fmaxf(chmax[i], sv);
      }
    }

    // online softmax per query row (rows live in 16-lane halves)
    #pragma unroll
    for (int i = 0; i < 8; ++i) {
      float v = chmax[i];
      v = fmaxf(v, __shfl_xor(v, 1, 32));
      v = fmaxf(v, __shfl_xor(v, 2, 32));
      v = fmaxf(v, __shfl_xor(v, 4, 32));
      v = fmaxf(v, __shfl_xor(v, 8, 32));
      float nm = fmaxf(rmax[i], v);
      float alpha = __expf(rmax[i] - nm);
      rmax[i] = nm;
      float csum = 0.f;
      #pragma unroll
      for (int nt = 0; nt < 4; ++nt) {
        float pv = __expf(sc[nt][i] - nm);
        sc[nt][i] = pv;
        csum += pv;
      }
      csum += __shfl_xor(csum, 1, 32);
      csum += __shfl_xor(csum, 2, 32);
      csum += __shfl_xor(csum, 4, 32);
      csum += __shfl_xor(csum, 8, 32);
      rsum[i] = rsum[i] * alpha + csum;
      #pragma unroll
      for (int dt = 0; dt < 8; ++dt) Oacc[dt][i] *= alpha;
    }

    // stage P (bf16) row-major for the next WMMA A operand
    #pragma unroll
    for (int nt = 0; nt < 4; ++nt)
      #pragma unroll
      for (int i = 0; i < 8; ++i)
        pt[(half * 8 + i) * 72 + nt * 16 + lrow] = f2bf(sc[nt][i]);

    // O += P * V
    #pragma unroll
    for (int s2 = 0; s2 < 2; ++s2) {
      BF16Tile Pa;
      const unsigned short* pp = pt + lrow * 72 + s2 * 32 + kbase;
      Pa.q[0] = *(const u32x4*)pp;
      Pa.q[1] = *(const u32x4*)(pp + 16);
      #pragma unroll
      for (int dt = 0; dt < 8; ++dt) {
        BF16Tile Vb;
        const unsigned short* vp = vTile + (dt * 16 + lrow) * 72 + s2 * 32 + kbase;
        Vb.q[0] = *(const u32x4*)vp;
        Vb.q[1] = *(const u32x4*)(vp + 16);
        Oacc[dt] = __builtin_amdgcn_wmma_f32_16x16x32_bf16(false, Pa.v, false, Vb.v,
                                                           (short)0, Oacc[dt], false, false);
      }
    }
  }

  // epilogue: out[b][p][n*128 + d]
  const int b = bn >> 2, n = bn & 3;
  #pragma unroll
  for (int i = 0; i < 8; ++i) {
    float inv = 1.f / rsum[i];
    int p = qbase + half * 8 + i;
    float* orow = out + ((size_t)b * HW + p) * 512 + n * DH;
    #pragma unroll
    for (int dt = 0; dt < 8; ++dt)
      orow[dt * 16 + lrow] = Oacc[dt][i] * inv;
  }
}

// ---------------------------------------------------------------------------
extern "C" void kernel_launch(void* const* d_in, const int* in_sizes, int n_in,
                              void* d_out, int out_size, void* d_ws, size_t ws_size,
                              hipStream_t stream) {
  (void)in_sizes; (void)n_in; (void)out_size; (void)ws_size;
  const float* x    = (const float*)d_in[0];   // (16, 512, 32, 32)
  const float* w    = (const float*)d_in[1];   // (1536, 512)
  const float* hrel = (const float*)d_in[2];   // (63, 128)
  const float* wrel = (const float*)d_in[3];   // (63, 128)
  float* out = (float*)d_out;

  // workspace layout (bf16 halves): Q, K, V^T (16 MB each), qw, qh (8 MB each) = 64 MB
  unsigned short* q_bf = (unsigned short*)d_ws;
  unsigned short* k_bf = q_bf + (size_t)BATCH * NH * HW * DH;
  unsigned short* v_t  = k_bf + (size_t)BATCH * NH * HW * DH;
  unsigned short* qwp  = v_t + (size_t)BATCH * NH * HW * DH;
  unsigned short* qhp  = qwp + (size_t)BATCH * NH * HW * 64;

  qkv_proj_kernel<<<dim3(8, 96, 16), 256, 0, stream>>>(x, w, q_bf, k_bf, v_t);
  rel_logits_kernel<<<dim3(8, 64), 256, 0, stream>>>(q_bf, hrel, wrel, qwp, qhp);
  attn_kernel<<<dim3(16, 64), 128, 0, stream>>>(q_bf, k_bf, v_t, qwp, qhp, out);
}
